// RayleighChannelLayer_5325759447645
// MI455X (gfx1250) — matrix operand order repaired and checked by reading, not compiled, not run on hardware
//
#include <hip/hip_runtime.h>
#include <math.h>

typedef __attribute__((ext_vector_type(16))) __bf16 v16bf;
typedef __attribute__((ext_vector_type(8)))  __bf16 v8bf;
typedef __attribute__((ext_vector_type(4)))  __bf16 v4bf;
typedef __attribute__((ext_vector_type(2)))  __bf16 v2bf;
typedef __attribute__((ext_vector_type(8)))  float  v8f;

static constexpr int kB   = 32;    // batch
static constexpr int kC   = 256;   // channels (M and K)
static constexpr int kHW  = 3136;  // 56*56 (N)
static constexpr int kMT  = 128;   // M tile per block (8 waves x 16)
static constexpr int kNT  = 112;   // N tile per block (7 x 16)
static constexpr int kKT  = 32;    // K step (one bf16 WMMA)
static constexpr int kLDK = 40;    // padded K stride in LDS: 80B, 16B-aligned, conflict-free
static constexpr int kTileA = kMT * kLDK;   // 5120 bf16 = 10240 B
static constexpr int kTileX = kNT * kLDK;   // 4480 bf16 =  8960 B

// f32 -> bf16, RNE. Native fptrunc: single v_cvt if the target has bf16 cvt
// instructions, otherwise clang's inline expansion (no libcall on amdgcn).
__device__ __forceinline__ __bf16 f2bf(float f) {
  return (__bf16)f;
}

// packed f32x2 -> bf16x2 (1 VALU op when v_cvt_pk_bf16_f32 is exposed)
__device__ __forceinline__ v2bf pk2(float a, float b) {
#if __has_builtin(__builtin_amdgcn_cvt_pk_bf16_f32)
  return __builtin_amdgcn_cvt_pk_bf16_f32(a, b);
#else
  v2bf r; r[0] = f2bf(a); r[1] = f2bf(b); return r;
#endif
}

__device__ __forceinline__ v4bf pk4(float a, float b, float c, float d) {
  v2bf lo = pk2(a, b), hi = pk2(c, d);
  return __builtin_shufflevector(lo, hi, 0, 1, 2, 3);
}

__device__ __forceinline__ v16bf cat8(v8bf lo, v8bf hi) {
  return __builtin_shufflevector(lo, hi, 0,1,2,3,4,5,6,7,8,9,10,11,12,13,14,15);
}

__global__ __launch_bounds__(256)
void rayleigh_wmma_kernel(const float* __restrict__ x,
                          const float* __restrict__ Hr,
                          const float* __restrict__ Hi,
                          const float* __restrict__ nzr,
                          const float* __restrict__ nzi,
                          float* __restrict__ out)
{
  __shared__ __bf16 sAr[2 * kTileA];
  __shared__ __bf16 sAi[2 * kTileA];
  __shared__ __bf16 sX [2 * kTileX];   // transposed tiles: [n][k]

  const int b    = blockIdx.z;
  const int mb   = blockIdx.y;              // 0..1   (M block of 128)
  const int nb   = blockIdx.x;              // 0..27  (N block of 112)
  const int tid  = threadIdx.x;             // 0..255
  const int lane = tid & 31;
  const int wave = tid >> 5;                // 0..7
  const int half = lane >> 4;               // 0/1
  const int l16  = lane & 15;

  const size_t hOff = (size_t)b * kC * kC + (size_t)(mb * kMT) * kC;
  const size_t xOff = (size_t)b * kC * kHW + (size_t)(nb * kNT);

  // per-thread fill geometry for the X tile
  const int xK  = tid >> 3;                 // 0..31
  const int xN0 = (tid & 7) * 14;           // 0,14,...,98

  auto loadA = [&](int kk, float4* fr, float4* fi) {
    #pragma unroll
    for (int j = 0; j < 4; ++j) {
      int idx = tid + 256 * j;              // (row, col4)
      int row = idx >> 3;
      int col = (idx & 7) << 2;
      fr[j] = *(const float4*)(Hr + hOff + (size_t)row * kC + kk + col);
      fi[j] = *(const float4*)(Hi + hOff + (size_t)row * kC + kk + col);
    }
  };
  auto storeA = [&](__bf16* dAr, __bf16* dAi, const float4* fr, const float4* fi) {
    #pragma unroll
    for (int j = 0; j < 4; ++j) {
      int idx = tid + 256 * j;
      int row = idx >> 3;
      int col = (idx & 7) << 2;
      *(v4bf*)&dAr[row * kLDK + col] = pk4(fr[j].x, fr[j].y, fr[j].z, fr[j].w);
      *(v4bf*)&dAi[row * kLDK + col] = pk4(fi[j].x, fi[j].y, fi[j].z, fi[j].w);
    }
  };
  auto loadX = [&](int kk, float2* g) {
    const float* p = x + xOff + (size_t)(kk + xK) * kHW + xN0;
    #pragma unroll
    for (int j = 0; j < 7; ++j) g[j] = *(const float2*)(p + 2 * j);
  };
  auto storeX = [&](__bf16* dX, const float2* g) {
    #pragma unroll
    for (int j = 0; j < 7; ++j) {
      v2bf p = pk2(g[j].x, g[j].y);
      dX[(xN0 + 2 * j + 0) * kLDK + xK] = p[0];
      dX[(xN0 + 2 * j + 1) * kLDK + xK] = p[1];
    }
  };

  v8f accR[7] = {};
  v8f accI[7] = {};

  // prologue: fill buffer 0
  {
    float4 r0[4], i0[4]; float2 x0[7];
    loadA(0, r0, i0);
    loadX(0, x0);
    storeA(sAr, sAi, r0, i0);
    storeX(sX, x0);
  }
  __syncthreads();

  #pragma unroll 2
  for (int it = 0; it < 8; ++it) {
    const int cur = it & 1;
    const int nxt = cur ^ 1;

    // stage next tile's global loads (overlaps the WMMAs below)
    float4 r1[4], i1[4]; float2 x1[7];
    if (it < 7) {
      loadA((it + 1) * kKT, r1, i1);
      loadX((it + 1) * kKT, x1);
    }

    // ---- compute from buffer `cur`
    const __bf16* Ar = sAr + cur * kTileA;
    const __bf16* Ai = sAi + cur * kTileA;
    const __bf16* Xb = sX  + cur * kTileX;

    const int am = wave * 16 + l16;
    // A 16x32 frag: half=0 -> K0..7|K16..23 ; half=1 -> K8..15|K24..31
    v16bf aR = cat8(*(const v8bf*)&Ar[am * kLDK + half * 8],
                    *(const v8bf*)&Ar[am * kLDK + 16 + half * 8]);
    v16bf aI = cat8(*(const v8bf*)&Ai[am * kLDK + half * 8],
                    *(const v8bf*)&Ai[am * kLDK + 16 + half * 8]);

    #pragma unroll
    for (int t = 0; t < 7; ++t) {
      const int bn = t * 16 + l16;
      // B 32x16 frag: half=0 -> K0..7|K16..23 ; half=1 -> K8..15|K24..31
      v16bf bb = cat8(*(const v8bf*)&Xb[bn * kLDK + half * 8],
                      *(const v8bf*)&Xb[bn * kLDK + 16 + half * 8]);
      accR[t] = __builtin_amdgcn_wmma_f32_16x16x32_bf16(false, aR, false, bb,
                                                        (short)0, accR[t], false, false);
      accI[t] = __builtin_amdgcn_wmma_f32_16x16x32_bf16(false, aI, false, bb,
                                                        (short)0, accI[t], false, false);
    }

    // ---- convert + store staged tile into the other buffer
    if (it < 7) {
      storeA(sAr + nxt * kTileA, sAi + nxt * kTileA, r1, i1);
      storeX(sX + nxt * kTileX, x1);
    }
    __syncthreads();
  }

  // ---- epilogue: scale, add noise, magnitude, store (C/D: vgpr r -> M = r + 8*half)
  const float scale = 0.044194173824159216f;  // 1/sqrt(2*256)
  const float nstd  = 0.01f;
  const int iBase = mb * kMT + wave * 16 + half * 8;
  const int nCol  = nb * kNT + l16;
  #pragma unroll
  for (int t = 0; t < 7; ++t) {
    #pragma unroll
    for (int r = 0; r < 8; ++r) {
      const int i = iBase + r;
      const size_t o = (size_t)(b * kC + i) * kHW + nCol + t * 16;
      float yr = accR[t][r] * scale + nzr[o] * nstd;
      float yi = accI[t][r] * scale + nzi[o] * nstd;
      out[o] = sqrtf(yr * yr + yi * yi);
    }
  }
}

extern "C" void kernel_launch(void* const* d_in, const int* in_sizes, int n_in,
                              void* d_out, int out_size, void* d_ws, size_t ws_size,
                              hipStream_t stream) {
  const float* x   = (const float*)d_in[0];
  const float* Hr  = (const float*)d_in[1];
  const float* Hi  = (const float*)d_in[2];
  const float* nzr = (const float*)d_in[3];
  const float* nzi = (const float*)d_in[4];
  float* out = (float*)d_out;

  dim3 grid(kHW / kNT, kC / kMT, kB);   // (28, 2, 32)
  rayleigh_wmma_kernel<<<grid, dim3(256), 0, stream>>>(x, Hr, Hi, nzr, nzi, out);
}
